// MH_137438953675
// MI455X (gfx1250) — matrix operand order repaired and checked by reading, not compiled, not run on hardware
//
#include <hip/hip_runtime.h>

// ===========================================================================
// MI455X / gfx1250 implementation.
//
// All dense math (1x1 convs, q/k/v/proj, attention score & PV products, MLP)
// runs through v_wmma_f32_16x16x32_bf16 (f32 accumulate).  f32 operands are
// staged into LDS as bf16 (RNE) with float4 global loads, and fragments are
// gathered as paired 32-bit LDS loads following the CDNA5 ISA 7.12.2 layouts.
// GEMM waves own a 32x32 output tile (4 WMMAs per K-step, A/B frags reused).
//
// Input flattening assumption (depth-first, dict insertion order):
//   0..2   x_1, x_2, x_3                        [B,L,D]
//   3..5   pos1, pos2, pos3                     [1,L,D]
//   6+10s  ssr{s}: bcdt.w, bcdt.b, dw_w, dw_b, hz.w, hz.b, out.w, out.b, A, D
//   36+30c chi{c}: n11.g,b, n12.g,b, n21.g,b, n23.g,b, n2.g,b,
//                  attn1.{q,k,v,proj}.{w,b}, attn2.{q,k,v,proj}.{w,b},
//                  fc1.w,b, fc2.w,b
//   126,127 norm.g, norm.b
//
// Workspace: 12 * B*L*D floats (~151 MB), reused across phases.
// ===========================================================================

static constexpr int  kB   = 64;
static constexpr int  kL   = 256;
static constexpr int  kD   = 192;
static constexpr int  kS   = 64;
static constexpr int  kNH  = 8;
static constexpr int  kHD  = 24;   // head dim (padded to 32 in LDS)
static constexpr int  kMLP = 768;
static constexpr long long kBLD = (long long)kB * kL * kD;   // 3,145,728
static constexpr long long kLD  = (long long)kL * kD;        // per-batch stride

typedef __attribute__((ext_vector_type(16))) __bf16 v16bf;
typedef __attribute__((ext_vector_type(8)))  float  v8f;

__device__ __forceinline__ __bf16 f2bf(float f) {
  unsigned u = __builtin_bit_cast(unsigned, f);
  unsigned r = u + 0x7FFFu + ((u >> 16) & 1u);           // round to nearest even
  unsigned short h = (unsigned short)(r >> 16);
  return __builtin_bit_cast(__bf16, h);
}

__device__ __forceinline__ v8f wmma_bf16(v16bf a, v16bf b, v8f c) {
  return __builtin_amdgcn_wmma_f32_16x16x32_bf16(false, a, false, b, (short)0, c,
                                                 false, false);
}

// --- Fragment gathers (paired bf16 -> 32-bit LDS loads) --------------------
// A layout (ISA 7.12.2): lane l -> row rb+(l&15); lanes<16 hold K {0..7,16..23},
// lanes>=16 hold K {8..15,24..31}.  Pairs (2j,2j+1) are adjacent in LDS.
__device__ __forceinline__ v16bf frag_a(const __bf16* T, int rb, int stride, int ks) {
  const int lane = threadIdx.x & 31;
  const int row = rb + (lane & 15);
  const int kh = (lane >> 4) * 4;                        // pair base 0 / 4
  const unsigned* p = (const unsigned*)(T + row * stride + ks);
  union { v16bf v; unsigned u[8]; } f;
#pragma unroll
  for (int j = 0; j < 8; ++j) f.u[j] = p[kh + (j < 4 ? j : j + 4)];
  return f.v;
}
// B layout: lane l -> column cb+(l&15); lanes<16 hold K {0..7,16..23}, lanes>=16
// hold K {8..15,24..31}.  T is column-major [col][k].
__device__ __forceinline__ v16bf frag_b(const __bf16* T, int cb, int stride, int ks) {
  const int lane = threadIdx.x & 31;
  const int col = cb + (lane & 15);
  const int kh = (lane >> 4) * 4;                        // pair base 0 / 4
  const unsigned* p = (const unsigned*)(T + col * stride + ks);
  union { v16bf v; unsigned u[8]; } f;
#pragma unroll
  for (int j = 0; j < 8; ++j) f.u[j] = p[kh + (j < 4 ? j : j + 4)];
  return f.v;
}

__device__ __forceinline__ void store_tile(float* C, int lcm, int lcn, int m0,
                                           int n0, const float* bias, int act,
                                           v8f acc) {
  const int lane = threadIdx.x & 31;
  const int n = n0 + (lane & 15);
  const int mb = m0 + ((lane >> 4) ? 8 : 0);
  const float bv = bias ? bias[n] : 0.0f;
#pragma unroll
  for (int p = 0; p < 8; ++p) {
    float v = acc[p] + bv;
    if (act == 1) v = 0.5f * v * (1.0f + erff(v * 0.70710678118654752f));
    C[(long long)(mb + p) * lcm + (long long)n * lcn] = v;
  }
}

// ---------------------------------------------------------------------------
// Generic strided batched GEMM:  C[m,n] = act( sum_k A[m,k]*B[k,n] + bias[n] )
// One wave per 32x32 tile (2x2 WMMA quadrants); K multiple of 32; M,N
// multiples of 32.  act: 0 = none, 1 = exact GELU.
// ---------------------------------------------------------------------------
__global__ void k_gemm(const float* __restrict__ A, int lam, int lak, long long sA,
                       const float* __restrict__ Bm, int lbk, int lbn, long long sB,
                       const float* __restrict__ bias,
                       float* __restrict__ C, int lcm, int lcn, long long sC,
                       int K, int act) {
  __shared__ __bf16 As[32 * 32];   // [row][k]
  __shared__ __bf16 Bt[32 * 32];   // [col(n)][k]  (transposed stage)
  const int lane = threadIdx.x;
  const int m0 = blockIdx.x * 32;
  const int n0 = blockIdx.y * 32;
  const int b  = blockIdx.z;
  A  += (long long)b * sA;
  Bm += (long long)b * sB;
  C  += (long long)b * sC;

  v8f a00 = {}, a01 = {}, a10 = {}, a11 = {};
  for (int k0 = 0; k0 < K; k0 += 32) {
    // ---- stage A (32 rows x 32 k) ----
    if (lak == 1 && (lam & 3) == 0) {            // K-contiguous rows
      const float* src = A + (long long)(m0 + lane) * lam + k0;
      __builtin_prefetch(src + 32, 0, 0);
#pragma unroll
      for (int j = 0; j < 8; ++j) {
        float4 f = *(const float4*)(src + 4 * j);
        __bf16* d = As + lane * 32 + 4 * j;
        d[0] = f2bf(f.x); d[1] = f2bf(f.y); d[2] = f2bf(f.z); d[3] = f2bf(f.w);
      }
    } else if (lam == 1 && (lak & 3) == 0) {     // M-contiguous columns
      const float* src = A + (long long)(k0 + lane) * lak + m0;
#pragma unroll
      for (int j = 0; j < 8; ++j) {
        float4 f = *(const float4*)(src + 4 * j);
        As[(4 * j + 0) * 32 + lane] = f2bf(f.x);
        As[(4 * j + 1) * 32 + lane] = f2bf(f.y);
        As[(4 * j + 2) * 32 + lane] = f2bf(f.z);
        As[(4 * j + 3) * 32 + lane] = f2bf(f.w);
      }
    } else {                                     // generic fallback
      for (int j = 0; j < 32; ++j)
        As[lane * 32 + j] =
            f2bf(A[(long long)(m0 + lane) * lam + (long long)(k0 + j) * lak]);
    }
    // ---- stage B transposed (32 n x 32 k) ----
    if (lbk == 1 && (lbn & 3) == 0) {            // K-contiguous per n
      const float* src = Bm + (long long)(n0 + lane) * lbn + k0;
      __builtin_prefetch(src + 32, 0, 0);
#pragma unroll
      for (int j = 0; j < 8; ++j) {
        float4 f = *(const float4*)(src + 4 * j);
        __bf16* d = Bt + lane * 32 + 4 * j;
        d[0] = f2bf(f.x); d[1] = f2bf(f.y); d[2] = f2bf(f.z); d[3] = f2bf(f.w);
      }
    } else if (lbn == 1 && (lbk & 3) == 0) {     // N-contiguous per k
      const float* src = Bm + (long long)(k0 + lane) * lbk + n0;
#pragma unroll
      for (int j = 0; j < 8; ++j) {
        float4 f = *(const float4*)(src + 4 * j);
        Bt[(4 * j + 0) * 32 + lane] = f2bf(f.x);
        Bt[(4 * j + 1) * 32 + lane] = f2bf(f.y);
        Bt[(4 * j + 2) * 32 + lane] = f2bf(f.z);
        Bt[(4 * j + 3) * 32 + lane] = f2bf(f.w);
      }
    } else {
      for (int j = 0; j < 32; ++j)
        Bt[lane * 32 + j] =
            f2bf(Bm[(long long)(k0 + j) * lbk + (long long)(n0 + lane) * lbn]);
    }
    __syncthreads();
    v16bf fa0 = frag_a(As, 0, 32, 0);
    v16bf fa1 = frag_a(As, 16, 32, 0);
    v16bf fb0 = frag_b(Bt, 0, 32, 0);
    v16bf fb1 = frag_b(Bt, 16, 32, 0);
    a00 = wmma_bf16(fa0, fb0, a00);
    a01 = wmma_bf16(fa0, fb1, a01);
    a10 = wmma_bf16(fa1, fb0, a10);
    a11 = wmma_bf16(fa1, fb1, a11);
    __syncthreads();
  }
  store_tile(C, lcm, lcn, m0,      n0,      bias, act, a00);
  store_tile(C, lcm, lcn, m0,      n0 + 16, bias, act, a01);
  store_tile(C, lcm, lcn, m0 + 16, n0,      bias, act, a10);
  store_tile(C, lcm, lcn, m0 + 16, n0 + 16, bias, act, a11);
}

// ---------------------------------------------------------------------------
// Fused cross-attention: one wave per (16-row Q tile, head, batch).
// q/k/v are [B,L,D] f32 (head h occupies columns h*24..h*24+23).
// Output written head-compacted into out[B,L,D].
// ---------------------------------------------------------------------------
__global__ void k_attn(const float* __restrict__ q, const float* __restrict__ k,
                       const float* __restrict__ v, float* __restrict__ outc) {
  __shared__ __bf16 Qs[16 * 32];    // [row][dim]
  __shared__ __bf16 Ks[256 * 32];   // [key][dim]
  __shared__ __bf16 Vt[32 * 256];   // [dim][key]  (transposed for PV B-frags)
  __shared__ float  Ssc[16 * 256];
  __shared__ __bf16 Pbf[16 * 256];
  __shared__ float  redA[32], redB[32];

  const int lane = threadIdx.x;
  const int m0 = blockIdx.x * 16;
  const int h  = blockIdx.y;
  const int b  = blockIdx.z;
  const float scale = 0.2041241452319315f;   // 24^-0.5
  const long long rowbase = (long long)b * kL * kD + h * kHD;

  {  // Q tile: lane pair (row, half) loads 12 floats = 3x float4, pre-scaled
    const int r = lane & 15, half = lane >> 4;
    const float* src = q + rowbase + (long long)(m0 + r) * kD + half * 12;
#pragma unroll
    for (int j = 0; j < 3; ++j) {
      float4 f = *(const float4*)(src + 4 * j);
      __bf16* d = Qs + r * 32 + half * 12 + 4 * j;
      d[0] = f2bf(f.x * scale); d[1] = f2bf(f.y * scale);
      d[2] = f2bf(f.z * scale); d[3] = f2bf(f.w * scale);
    }
    if (half) {
      __bf16* d = Qs + r * 32 + 24;
#pragma unroll
      for (int j = 0; j < 8; ++j) d[j] = f2bf(0.0f);
    }
  }
  for (int r = lane; r < kL; r += 32) {  // K rows + V columns (transposed)
    const float* ksrc = k + rowbase + (long long)r * kD;
    const float* vsrc = v + rowbase + (long long)r * kD;
#pragma unroll
    for (int j = 0; j < 6; ++j) {
      float4 f = *(const float4*)(ksrc + 4 * j);
      __bf16* d = Ks + r * 32 + 4 * j;
      d[0] = f2bf(f.x); d[1] = f2bf(f.y); d[2] = f2bf(f.z); d[3] = f2bf(f.w);
      float4 g = *(const float4*)(vsrc + 4 * j);
      Vt[(4 * j + 0) * kL + r] = f2bf(g.x);
      Vt[(4 * j + 1) * kL + r] = f2bf(g.y);
      Vt[(4 * j + 2) * kL + r] = f2bf(g.z);
      Vt[(4 * j + 3) * kL + r] = f2bf(g.w);
    }
#pragma unroll
    for (int c = 24; c < 32; ++c) {
      Ks[r * 32 + c] = f2bf(0.0f);
      Vt[c * kL + r] = f2bf(0.0f);
    }
  }
  __syncthreads();

  const int col = lane & 15;
  const int hi  = lane >> 4;
  v16bf aq = frag_a(Qs, 0, 32, 0);
  for (int nt = 0; nt < 16; ++nt) {            // scores: 16 x 256
    v16bf bk = frag_b(Ks, nt * 16, 32, 0);
    v8f sc = {};
    sc = wmma_bf16(aq, bk, sc);
    const int mb = hi * 8;
#pragma unroll
    for (int p = 0; p < 8; ++p) Ssc[(mb + p) * kL + nt * 16 + col] = sc[p];
  }
  __syncthreads();

  {                                            // row softmax over 256 keys
    const int row = lane & 15;
    const int c0 = hi * 128;
    float m = -3.0e38f;
    for (int c = 0; c < 128; ++c) m = fmaxf(m, Ssc[row * kL + c0 + c]);
    redA[lane] = m;
    __syncthreads();
    m = fmaxf(redA[row], redA[row + 16]);
    float s = 0.0f;
    for (int c = 0; c < 128; ++c) {
      float e = __expf(Ssc[row * kL + c0 + c] - m);
      Ssc[row * kL + c0 + c] = e;
      s += e;
    }
    redB[lane] = s;
    __syncthreads();
    float inv = 1.0f / (redB[row] + redB[row + 16]);
    for (int c = 0; c < 128; ++c)
      Pbf[row * kL + c0 + c] = f2bf(Ssc[row * kL + c0 + c] * inv);
  }
  __syncthreads();

  for (int nt = 0; nt < 2; ++nt) {             // O = P (16x256) @ V (256x32)
    v8f acc = {};
    for (int ks = 0; ks < kL; ks += 32) {
      v16bf ap = frag_a(Pbf, 0, kL, ks);
      v16bf bv = frag_b(Vt, nt * 16, kL, ks);
      acc = wmma_bf16(ap, bv, acc);
    }
    const int oc = nt * 16 + col;
    if (oc < kHD) {                            // post-WMMA divergence only
      const int mb = hi * 8;
#pragma unroll
      for (int p = 0; p < 8; ++p)
        outc[rowbase + (long long)(m0 + mb + p) * kD + oc] = acc[p];
    }
  }
}

// --------------------------- elementwise kernels ---------------------------

__global__ void k_add_pos(const float* __restrict__ x, const float* __restrict__ pos,
                          float* __restrict__ out) {
  long long i = (long long)blockIdx.x * blockDim.x + threadIdx.x;
  if (i >= kBLD) return;
  int ld = (int)(i % kLD);
  out[i] = x[i] + pos[ld];
}

__global__ void k_dwconv(const float* __restrict__ in, const float* __restrict__ w,
                         const float* __restrict__ bias, float* __restrict__ out) {
  long long i = (long long)blockIdx.x * blockDim.x + threadIdx.x;
  const long long total = (long long)kB * 3 * kS * kL;
  if (i >= total) return;
  const int xw = (int)(i & 15);
  const int yh = (int)((i >> 4) & 15);
  const int c  = (int)((i >> 8) % (3 * kS));
  const long long base = (i >> 8) << 8;
  float acc = bias[c];
#pragma unroll
  for (int dy = 0; dy < 3; ++dy) {
    int yy = yh + dy - 1;
    if (yy < 0 || yy > 15) continue;
#pragma unroll
    for (int dx = 0; dx < 3; ++dx) {
      int xx = xw + dx - 1;
      if (xx < 0 || xx > 15) continue;
      acc += in[base + yy * 16 + xx] * w[c * 9 + dy * 3 + dx];
    }
  }
  out[i] = acc;
}

// A = softmax(dt + Ap) over L; AB = A * Bm.  bcdt2 = [B, 3S, L].
__global__ void k_softmax_ab(const float* __restrict__ bcdt2,
                             const float* __restrict__ Ap, float* __restrict__ AB) {
  __shared__ float red[256];
  const int s2 = blockIdx.x, b = blockIdx.y, l = threadIdx.x;
  const long long base = (long long)b * 3 * kS * kL;
  float val = bcdt2[base + (long long)(2 * kS + s2) * kL + l] + Ap[s2];
  red[l] = val;  __syncthreads();
  for (int off = 128; off > 0; off >>= 1) {
    if (l < off) red[l] = fmaxf(red[l], red[l + off]);
    __syncthreads();
  }
  float m = red[0];  __syncthreads();
  float e = __expf(val - m);
  red[l] = e;  __syncthreads();
  for (int off = 128; off > 0; off >>= 1) {
    if (l < off) red[l] += red[l + off];
    __syncthreads();
  }
  float sum = red[0];
  float Bv = bcdt2[base + (long long)s2 * kL + l];
  AB[((long long)b * kS + s2) * kL + l] = (e / sum) * Bv;
}

// g = h1*silu(z) + h1*Dp ; hz = [B, 2D, S]
__global__ void k_gate(const float* __restrict__ hz, const float* __restrict__ Dp,
                       float* __restrict__ g) {
  long long i = (long long)blockIdx.x * blockDim.x + threadIdx.x;
  const long long total = (long long)kB * kD * kS;
  if (i >= total) return;
  const int s = (int)(i % kS);
  const long long t = i / kS;
  const int d = (int)(t % kD);
  const int b = (int)(t / kD);
  const long long hb = (long long)b * 2 * kD * kS;
  float h1 = hz[hb + (long long)d * kS + s];
  float z  = hz[hb + (long long)(kD + d) * kS + s];
  float sig = 1.0f / (1.0f + __expf(-z));
  g[i] = h1 * z * sig + h1 * Dp[0];
}

__global__ void k_ln(const float* __restrict__ x, const float* __restrict__ g,
                     const float* __restrict__ bta, float* __restrict__ out) {
  __shared__ float red[256];
  const long long row = blockIdx.x;
  const int t = threadIdx.x;
  float v = (t < kD) ? x[row * kD + t] : 0.0f;
  red[t] = v;  __syncthreads();
  for (int off = 128; off > 0; off >>= 1) {
    if (t < off) red[t] += red[t + off];
    __syncthreads();
  }
  float mean = red[0] * (1.0f / (float)kD);  __syncthreads();
  float dv = (t < kD) ? (v - mean) : 0.0f;
  red[t] = dv * dv;  __syncthreads();
  for (int off = 128; off > 0; off >>= 1) {
    if (t < off) red[t] += red[t + off];
    __syncthreads();
  }
  float inv = rsqrtf(red[0] * (1.0f / (float)kD) + 1e-6f);
  if (t < kD) out[row * kD + t] = dv * inv * g[t] + bta[t];
}

__global__ void k_add3(const float* __restrict__ a, const float* __restrict__ b,
                       const float* __restrict__ c, float* __restrict__ o, long long n) {
  long long i = (long long)blockIdx.x * blockDim.x + threadIdx.x;
  if (i < n) o[i] = a[i] + b[i] + c[i];
}

__global__ void k_add2(const float* __restrict__ a, const float* __restrict__ b,
                       float* __restrict__ o, long long n) {
  long long i = (long long)blockIdx.x * blockDim.x + threadIdx.x;
  if (i < n) o[i] = a[i] + b[i];
}

// Final: LN(concat(x1,x2,x3)) -> out[:, :768, :]; plus raw x1,x2,x3 copies.
__global__ void k_final(const float* __restrict__ x1, const float* __restrict__ x2,
                        const float* __restrict__ x3, const float* __restrict__ g,
                        const float* __restrict__ bta, float* __restrict__ dout) {
  __shared__ float red[256];
  const int l = blockIdx.x, strm = blockIdx.y, b = blockIdx.z, t = threadIdx.x;
  const float* src = (strm == 0) ? x1 : (strm == 1) ? x2 : x3;
  const long long row = (long long)b * kL + l;
  float v = (t < kD) ? src[row * kD + t] : 0.0f;
  red[t] = v;  __syncthreads();
  for (int off = 128; off > 0; off >>= 1) {
    if (t < off) red[t] += red[t + off];
    __syncthreads();
  }
  float mean = red[0] * (1.0f / (float)kD);  __syncthreads();
  float dv = (t < kD) ? (v - mean) : 0.0f;
  red[t] = dv * dv;  __syncthreads();
  for (int off = 128; off > 0; off >>= 1) {
    if (t < off) red[t] += red[t + off];
    __syncthreads();
  }
  float inv = rsqrtf(red[0] * (1.0f / (float)kD) + 1e-6f);
  if (t < kD) {
    const long long OUT0 = (long long)kB * 768 * kD;
    dout[((long long)b * 768 + (long long)strm * kL + l) * kD + t] =
        dv * inv * g[t] + bta[t];
    dout[OUT0 + (long long)strm * kBLD + row * kD + t] = v;
  }
}

// ===========================================================================
// Host orchestration
// ===========================================================================
extern "C" void kernel_launch(void* const* d_in, const int* in_sizes, int n_in,
                              void* d_out, int out_size, void* d_ws, size_t ws_size,
                              hipStream_t stream) {
  (void)in_sizes; (void)n_in; (void)out_size; (void)ws_size;
  auto P = [&](int i) -> const float* { return (const float*)d_in[i]; };

  float* ws = (float*)d_ws;
  float* st1 = ws;
  float* st2 = ws + kBLD;
  float* st3 = ws + 2 * kBLD;
  float* pool = ws + 3 * kBLD;            // 9*kBLD reusable scratch
  // phase-1 (HSMSSD) views
  float* Tx    = pool;
  float* bcdt  = pool + kBLD;
  float* bcdt2 = pool + 2 * kBLD;
  float* ABb   = pool + 3 * kBLD;
  float* hbuf  = ABb + (long long)kB * kS * kL;
  float* hzbuf = hbuf + (long long)kB * kD * kS;
  float* gbuf  = hzbuf + (long long)kB * 2 * kD * kS;
  float* h2buf = gbuf + (long long)kB * kD * kS;
  // phase-2 (chi) views
  float* lnA  = pool;
  float* lnB  = pool + kBLD;
  float* qb   = pool + 2 * kBLD;
  float* kbuf = pool + 3 * kBLD;
  float* vbuf = pool + 4 * kBLD;
  float* cmp  = pool + 5 * kBLD;
  float* abuf = pool + 6 * kBLD;
  float* bbuf = pool + 7 * kBLD;
  float* xbuf = pool + 8 * kBLD;
  float* m1   = qb;                       // 4*kBLD span qb..cmp == B*L*MLP

  auto EW = [](long long n) { return dim3((unsigned)((n + 255) / 256)); };

  auto gemm = [&](const float* A, int lam, int lak, long long sA,
                  const float* Bm, int lbk, int lbn, long long sB,
                  const float* bias, float* C, int lcm, int lcn, long long sC,
                  int M, int N, int K, int act) {
    k_gemm<<<dim3(M / 32, N / 32, kB), dim3(32), 0, stream>>>(
        A, lam, lak, sA, Bm, lbk, lbn, sB, bias, C, lcm, lcn, sC, K, act);
  };
  auto ln = [&](const float* x, const float* g, const float* b, float* out) {
    k_ln<<<dim3(kB * kL), 256, 0, stream>>>(x, g, b, out);
  };

  // ------------------------- HSMSSD per stream -------------------------
  auto hsm = [&](int sidx, float* state) {
    const int pb = 6 + 10 * sidx;
    k_add_pos<<<EW(kBLD), 256, 0, stream>>>(P(sidx), P(3 + sidx), Tx);
    // bcdt[b,o,l] = sum_i W[o,i]*Tx[b,l,i] + b[o]   (M=l, N=o, K=i)
    gemm(Tx, kD, 1, kLD,  P(pb + 0), 1, kD, 0,  P(pb + 1),
         bcdt, 1, kL, (long long)3 * kS * kL,  kL, 3 * kS, kD, 0);
    k_dwconv<<<EW((long long)kB * 3 * kS * kL), 256, 0, stream>>>(
        bcdt, P(pb + 2), P(pb + 3), bcdt2);
    k_softmax_ab<<<dim3(kS, kB), 256, 0, stream>>>(bcdt2, P(pb + 8), ABb);
    // h[b,d,s] = sum_l Tx[b,l,d]*AB[b,s,l]   (M=d, N=s, K=l)
    gemm(Tx, 1, kD, kLD,  ABb, 1, kL, (long long)kS * kL,  nullptr,
         hbuf, kS, 1, (long long)kD * kS,  kD, kS, kL, 0);
    // hz[b,o,s] = sum_d Whz[o,d]*h[b,d,s] + b[o]   (M=s, N=o, K=d)
    gemm(hbuf, 1, kS, (long long)kD * kS,  P(pb + 4), 1, kD, 0,  P(pb + 5),
         hzbuf, 1, kS, (long long)2 * kD * kS,  kS, 2 * kD, kD, 0);
    k_gate<<<EW((long long)kB * kD * kS), 256, 0, stream>>>(hzbuf, P(pb + 9), gbuf);
    // h2[b,o,s] = sum_d Wout[o,d]*g[b,d,s] + b[o]  (M=s, N=o, K=d)
    gemm(gbuf, 1, kS, (long long)kD * kS,  P(pb + 6), 1, kD, 0,  P(pb + 7),
         h2buf, 1, kS, (long long)kD * kS,  kS, kD, kD, 0);
    // state[b,l,d] = sum_s h2[b,d,s]*Cm[b,s,l]   (M=d, N=l, K=s)
    gemm(h2buf, kS, 1, (long long)kD * kS,
         bcdt2 + (long long)kS * kL, kL, 1, (long long)3 * kS * kL,
         nullptr, state, 1, kD, kLD,  kD, kL, kS, 0);
  };

  hsm(0, st1);  hsm(1, st2);  hsm(2, st3);

  // ----------------------------- chi blocks -----------------------------
  auto cross = [&](const float* srcq, const float* srckv, int pw, float* dst) {
    gemm(srcq, kD, 1, kLD, P(pw + 0), 1, kD, 0, P(pw + 1), qb,  kD, 1, kLD, kL, kD, kD, 0);
    gemm(srckv, kD, 1, kLD, P(pw + 2), 1, kD, 0, P(pw + 3), kbuf, kD, 1, kLD, kL, kD, kD, 0);
    gemm(srckv, kD, 1, kLD, P(pw + 4), 1, kD, 0, P(pw + 5), vbuf, kD, 1, kLD, kL, kD, kD, 0);
    k_attn<<<dim3(kL / 16, kNH, kB), dim3(32), 0, stream>>>(qb, kbuf, vbuf, cmp);
    gemm(cmp, kD, 1, kLD, P(pw + 6), 1, kD, 0, P(pw + 7), dst, kD, 1, kLD, kL, kD, kD, 0);
  };
  auto chi = [&](float* xi, const float* xj, const float* xk, int pb) {
    ln(xi, P(pb + 0), P(pb + 1), lnA);
    ln(xj, P(pb + 2), P(pb + 3), lnB);
    cross(lnA, lnB, pb + 10, abuf);
    ln(xi, P(pb + 4), P(pb + 5), lnA);
    ln(xk, P(pb + 6), P(pb + 7), lnB);
    cross(lnA, lnB, pb + 18, bbuf);
    k_add3<<<EW(kBLD), 256, 0, stream>>>(xi, abuf, bbuf, xbuf, kBLD);
    ln(xbuf, P(pb + 8), P(pb + 9), lnA);
    gemm(lnA, kD, 1, kLD, P(pb + 26), 1, kD, 0, P(pb + 27),
         m1, kMLP, 1, (long long)kL * kMLP, kL, kMLP, kD, 1);      // GELU
    gemm(m1, kMLP, 1, (long long)kL * kMLP, P(pb + 28), 1, kMLP, 0, P(pb + 29),
         lnB, kD, 1, kLD, kL, kD, kMLP, 0);
    k_add2<<<EW(kBLD), 256, 0, stream>>>(xbuf, lnB, xi, kBLD);
  };

  chi(st1, st2, st3, 36);
  chi(st2, st1, st3, 66);
  chi(st3, st2, st1, 96);

  // -------------------------- final LN + copies --------------------------
  k_final<<<dim3(kL, 3, kB), 256, 0, stream>>>(st1, st2, st3, P(126), P(127),
                                               (float*)d_out);
}
